// EfficientAtt_73220602462448
// MI455X (gfx1250) — matrix operand order repaired and checked by reading, not compiled, not run on hardware
//
#include <hip/hip_runtime.h>

// ---------------------------------------------------------------------------
// MI455X / gfx1250 implementation: bf16 WMMA everywhere, wave32, WGP-friendly
// 256-thread (8-wave) blocks for the GEMMs.
// ---------------------------------------------------------------------------

typedef __attribute__((ext_vector_type(16))) __bf16 v16bf;
typedef __attribute__((ext_vector_type(8)))  float  v8f;

union BFrag { v16bf v; uint4 u[2]; };

// ------------------------- prep kernels ------------------------------------

__global__ void k_convert(const float* __restrict__ src, __bf16* __restrict__ dst, int n) {
  int i = blockIdx.x * blockDim.x + threadIdx.x;
  if (i < n) dst[i] = (__bf16)src[i];
}

// BN (eval, mean 0 / var 1) + hardswish, fused with f32->bf16
__global__ void k_act(const float* __restrict__ x, const float* __restrict__ g,
                      const float* __restrict__ bt, __bf16* __restrict__ xg) {
  int i = blockIdx.x * 256 + threadIdx.x;
  int c = i & 255;
  const float inv_std = 0.99999500003749969f;  // 1/sqrt(1+1e-5)
  float xb = x[i] * (g[c] * inv_std) + bt[c];
  float t = fminf(fmaxf(xb + 3.0f, 0.0f), 6.0f);
  xg[i] = (__bf16)(xb * t * (1.0f / 6.0f));
}

// ------------------------- QKV GEMM (M=32768,N=768,K=256) -------------------
// 128x128 tile per block; 8 waves, each wave = 32x64 (2x4 WMMA tiles).

__global__ __launch_bounds__(256) void k_gemm_qkv(
    const __bf16* __restrict__ A, const __bf16* __restrict__ Bw,
    __bf16* __restrict__ qo, __bf16* __restrict__ ko, __bf16* __restrict__ vo) {
  __shared__ alignas(16) __bf16 As[128][40];
  __shared__ alignas(16) __bf16 Bs[128][40];
  const int tid  = threadIdx.x;
  const int lane = tid & 31;
  const int w    = tid >> 5;
  const int rowBase = blockIdx.x * 128;
  const int colBase = blockIdx.y * 128;
  const int wm = (w & 3) * 32;
  const int wn = (w >> 2) * 64;
  v8f acc[2][4] = {};
  for (int kt = 0; kt < 256; kt += 32) {
    for (int ld = 0; ld < 2; ++ld) {
      int id = tid + ld * 256;
      int r = id >> 2, cp = (id & 3) * 8;
      *(uint4*)&As[r][cp] = *(const uint4*)(A  + (size_t)(rowBase + r) * 256 + kt + cp);
      *(uint4*)&Bs[r][cp] = *(const uint4*)(Bw + (size_t)(colBase + r) * 256 + kt + cp);
    }
    if (kt + 32 < 256) {
      __builtin_prefetch(A  + (size_t)(rowBase + (tid >> 1)) * 256 + kt + 32, 0, 1);
      __builtin_prefetch(Bw + (size_t)(colBase + (tid >> 1)) * 256 + kt + 32, 0, 1);
    }
    __syncthreads();
    const int kb  = (lane & 16) ? 8 : 0;    // A: lanes 16-31 carry K=8..15 / 24..31
    const int kb2 = (lane & 16) ? 8 : 0;    // B: lanes 16-31 carry K=8..15 / 24..31
    BFrag fa[2], fb[4];
    for (int a = 0; a < 2; ++a) {
      int m = wm + a * 16 + (lane & 15);
      fa[a].u[0] = *(const uint4*)&As[m][kb];
      fa[a].u[1] = *(const uint4*)&As[m][kb + 16];
    }
    for (int bi = 0; bi < 4; ++bi) {
      int n = wn + bi * 16 + (lane & 15);
      fb[bi].u[0] = *(const uint4*)&Bs[n][kb2];
      fb[bi].u[1] = *(const uint4*)&Bs[n][kb2 + 16];
    }
    for (int a = 0; a < 2; ++a)
      for (int bi = 0; bi < 4; ++bi)
        acc[a][bi] = __builtin_amdgcn_wmma_f32_16x16x32_bf16(
            false, fa[a].v, false, fb[bi].v, (short)0, acc[a][bi], false, false);
    __syncthreads();
  }
  const int hi8 = (lane >> 4) << 3;
  for (int a = 0; a < 2; ++a)
    for (int bi = 0; bi < 4; ++bi) {
      int gcol = colBase + wn + bi * 16 + (lane & 15);
      int which = gcol >> 8, rest = gcol & 255;       // uniform per 16-col tile
      __bf16* dst = (which == 0) ? qo : (which == 1) ? ko : vo;
      for (int r = 0; r < 8; ++r) {
        int m = rowBase + wm + a * 16 + r + hi8;
        dst[(size_t)m * 256 + rest] = (__bf16)acc[a][bi][r];
      }
    }
}

// ------------------- v [n][c] -> channel-major image [b][c][n] --------------

__global__ __launch_bounds__(256) void k_vimg(const __bf16* __restrict__ v,
                                              __bf16* __restrict__ vimg) {
  __shared__ __bf16 tb[64][65];
  int b = blockIdx.z, n0 = blockIdx.x * 64, c0 = blockIdx.y * 64;
  int t = threadIdx.x;
  for (int i = 0; i < 16; ++i) {
    int id = t + i * 256;
    int rn = id >> 6, cc = id & 63;
    tb[rn][cc] = v[(size_t)(b * 4096 + n0 + rn) * 256 + c0 + cc];
  }
  __syncthreads();
  for (int i = 0; i < 16; ++i) {
    int id = t + i * 256;
    int cc = id >> 6, rn = id & 63;
    vimg[(size_t)(b * 256 + c0 + cc) * 4096 + n0 + rn] = tb[rn][cc];
  }
}

// --------------- softmax stats over token axis per (b, channel) -------------

__global__ __launch_bounds__(256) void k_softmax(const __bf16* __restrict__ kq,
                                                 float* __restrict__ smax,
                                                 float* __restrict__ sinv) {
  __shared__ float red[256];
  int g = blockIdx.x, b = g >> 8, c = g & 255;
  int t = threadIdx.x;
  float vals[16];
  float mx = -3.0e38f;
  size_t base = (size_t)b * 4096 * 256 + c;
  for (int i = 0; i < 16; ++i) {
    vals[i] = (float)kq[base + (size_t)(t + i * 256) * 256];
    mx = fmaxf(mx, vals[i]);
  }
  red[t] = mx; __syncthreads();
  for (int s = 128; s > 0; s >>= 1) {
    if (t < s) red[t] = fmaxf(red[t], red[t + s]);
    __syncthreads();
  }
  mx = red[0];
  __syncthreads();
  float sum = 0.f;
  for (int i = 0; i < 16; ++i) sum += __expf(vals[i] - mx);
  red[t] = sum; __syncthreads();
  for (int s = 128; s > 0; s >>= 1) {
    if (t < s) red[t] += red[t + s];
    __syncthreads();
  }
  if (t == 0) { smax[g] = mx; sinv[g] = 1.0f / red[0]; }
}

// ---- kv[k,v] = sum_n softmax(k)[n,k] * v[n,v] per (b,h); stores kv^T*scale --

__global__ __launch_bounds__(128) void k_kv(const __bf16* __restrict__ kq,
                                            const __bf16* __restrict__ vq,
                                            const float* __restrict__ smax,
                                            const float* __restrict__ sinv,
                                            __bf16* __restrict__ kvT) {
  __shared__ alignas(16) __bf16 ksmT[32][48];  // [k-ch][token]
  __shared__ alignas(16) __bf16 vT[32][48];    // [v-ch][token]
  __shared__ float smc[32], sic[32];
  int bh = blockIdx.x, b = bh >> 3, h = bh & 7;
  int t = threadIdx.x, lane = t & 31, w = t >> 5;
  if (t < 32) { smc[t] = smax[b * 256 + h * 32 + t]; sic[t] = sinv[b * 256 + h * 32 + t]; }
  __syncthreads();
  int mt = (w >> 1) * 16, nt = (w & 1) * 16;
  int c = t & 31, t4 = t >> 5;
  v8f acc = {};
  for (int it = 0; it < 128; ++it) {
    int tbase = it * 32;
    for (int s = 0; s < 8; ++s) {
      int tok = t4 * 8 + s;
      size_t idx = (size_t)(b * 4096 + tbase + tok) * 256 + h * 32 + c;
      float kf = (float)kq[idx];
      ksmT[c][tok] = (__bf16)(__expf(kf - smc[c]) * sic[c]);
      vT[c][tok]   = vq[idx];
    }
    __syncthreads();
    const int kb  = (lane & 16) ? 8 : 0;
    const int kb2 = (lane & 16) ? 8 : 0;
    BFrag fa, fb;
    int m = mt + (lane & 15), n = nt + (lane & 15);
    fa.u[0] = *(const uint4*)&ksmT[m][kb];
    fa.u[1] = *(const uint4*)&ksmT[m][kb + 16];
    fb.u[0] = *(const uint4*)&vT[n][kb2];
    fb.u[1] = *(const uint4*)&vT[n][kb2 + 16];
    acc = __builtin_amdgcn_wmma_f32_16x16x32_bf16(false, fa.v, false, fb.v,
                                                  (short)0, acc, false, false);
    __syncthreads();
  }
  const float scale = 0.17677669529663689f;  // 32^-0.5 folded into kv
  int vch = nt + (lane & 15), hi8 = (lane >> 4) << 3;
  for (int r = 0; r < 8; ++r) {
    int kch = mt + r + hi8;
    kvT[(size_t)bh * 1024 + vch * 32 + kch] = (__bf16)(acc[r] * scale);
  }
}

// ---- fused: eff = q@kv (WMMA) + crpe = q * dwconv(v); attn -> bf16 ---------

template <int S>
__global__ __launch_bounds__(256) void k_effcrpe(
    const __bf16* __restrict__ qb, const __bf16* __restrict__ vimg,
    const __bf16* __restrict__ kvT, const float* __restrict__ wsrc,
    const float* __restrict__ bsrc, __bf16* __restrict__ attn, int hbase) {
  __shared__ alignas(16) __bf16 vtile[32][640];   // 32 ch x (10 rows x 64)
  __shared__ alignas(16) __bf16 qtile[256][32];   // 256 tokens x 32 ch
  __shared__ float wts[32][S * S];
  __shared__ float bias_s[32];
  int yblk = blockIdx.x, h = hbase + blockIdx.y, b = blockIdx.z;
  int c0 = h * 32, y0 = yblk * 4;
  int t = threadIdx.x, lane = t & 31, w = t >> 5;
  int cgrp = (h - hbase) * 32;  // channel index within this conv group
  for (int id = t; id < 32 * S * S; id += 256) {
    int cc = id / (S * S), kk = id % (S * S);
    wts[cc][kk] = wsrc[(size_t)(cgrp + cc) * (S * S) + kk];
  }
  if (t < 32) bias_s[t] = bsrc[cgrp + t];
  for (int id = t; id < 32 * 640; id += 256) {
    int cc = id / 640, rem = id % 640;
    int ry = rem >> 6, xx = rem & 63;
    int gy = y0 - 3 + ry;
    __bf16 val = (__bf16)0.0f;
    if (gy >= 0 && gy < 64)
      val = vimg[(size_t)(b * 256 + c0 + cc) * 4096 + gy * 64 + xx];
    vtile[cc][rem] = val;
  }
  for (int id = t; id < 256 * 32; id += 256) {
    int tok = id >> 5, cc = id & 31;
    int gn = yblk * 256 + tok;
    qtile[tok][cc] = qb[(size_t)(b * 4096 + gn) * 256 + c0 + cc];
  }
  __syncthreads();
  const int kb  = (lane & 16) ? 8 : 0;
  const int kb2 = (lane & 16) ? 8 : 0;
  const int bh  = b * 8 + h;
  for (int i = 0; i < 4; ++i) {
    int tile = w * 4 + i;
    int tokT = (tile >> 1) * 16, chT = (tile & 1) * 16;
    BFrag fa, fb;
    int m = tokT + (lane & 15);
    fa.u[0] = *(const uint4*)&qtile[m][kb];
    fa.u[1] = *(const uint4*)&qtile[m][kb + 16];
    int ch = chT + (lane & 15);
    const __bf16* kvp = kvT + (size_t)bh * 1024 + ch * 32 + kb2;
    fb.u[0] = *(const uint4*)kvp;
    fb.u[1] = *(const uint4*)(kvp + 16);
    v8f acc = {};
    acc = __builtin_amdgcn_wmma_f32_16x16x32_bf16(false, fa.v, false, fb.v,
                                                  (short)0, acc, false, false);
    int hi8 = (lane >> 4) << 3;
    for (int r = 0; r < 8; ++r) {
      int tok = tokT + r + hi8;
      int xx = tok & 63, dy = tok >> 6;
      float conv = bias_s[ch];
#pragma unroll
      for (int ki = 0; ki < S; ++ki) {
        int yy = dy + 3 + ki - S / 2;  // always in [0,10)
#pragma unroll
        for (int kj = 0; kj < S; ++kj) {
          int xj = xx + kj - S / 2;
          if (xj >= 0 && xj < 64)
            conv += wts[ch][ki * S + kj] * (float)vtile[ch][yy * 64 + xj];
        }
      }
      float qv = (float)qtile[tok][ch];
      int gn = yblk * 256 + tok;
      attn[(size_t)(b * 4096 + gn) * 256 + c0 + ch] = (__bf16)(acc[r] + qv * conv);
    }
  }
}

// ------------------------- proj GEMM (N=256) + bias -------------------------

__global__ __launch_bounds__(256) void k_gemm_proj(
    const __bf16* __restrict__ A, const __bf16* __restrict__ Bw,
    const float* __restrict__ pb, float* __restrict__ out) {
  __shared__ alignas(16) __bf16 As[128][40];
  __shared__ alignas(16) __bf16 Bs[128][40];
  const int tid  = threadIdx.x;
  const int lane = tid & 31;
  const int w    = tid >> 5;
  const int rowBase = blockIdx.x * 128;
  const int colBase = blockIdx.y * 128;
  const int wm = (w & 3) * 32;
  const int wn = (w >> 2) * 64;
  v8f acc[2][4] = {};
  for (int kt = 0; kt < 256; kt += 32) {
    for (int ld = 0; ld < 2; ++ld) {
      int id = tid + ld * 256;
      int r = id >> 2, cp = (id & 3) * 8;
      *(uint4*)&As[r][cp] = *(const uint4*)(A  + (size_t)(rowBase + r) * 256 + kt + cp);
      *(uint4*)&Bs[r][cp] = *(const uint4*)(Bw + (size_t)(colBase + r) * 256 + kt + cp);
    }
    if (kt + 32 < 256) {
      __builtin_prefetch(A  + (size_t)(rowBase + (tid >> 1)) * 256 + kt + 32, 0, 1);
      __builtin_prefetch(Bw + (size_t)(colBase + (tid >> 1)) * 256 + kt + 32, 0, 1);
    }
    __syncthreads();
    const int kb  = (lane & 16) ? 8 : 0;
    const int kb2 = (lane & 16) ? 8 : 0;
    BFrag fa[2], fb[4];
    for (int a = 0; a < 2; ++a) {
      int m = wm + a * 16 + (lane & 15);
      fa[a].u[0] = *(const uint4*)&As[m][kb];
      fa[a].u[1] = *(const uint4*)&As[m][kb + 16];
    }
    for (int bi = 0; bi < 4; ++bi) {
      int n = wn + bi * 16 + (lane & 15);
      fb[bi].u[0] = *(const uint4*)&Bs[n][kb2];
      fb[bi].u[1] = *(const uint4*)&Bs[n][kb2 + 16];
    }
    for (int a = 0; a < 2; ++a)
      for (int bi = 0; bi < 4; ++bi)
        acc[a][bi] = __builtin_amdgcn_wmma_f32_16x16x32_bf16(
            false, fa[a].v, false, fb[bi].v, (short)0, acc[a][bi], false, false);
    __syncthreads();
  }
  const int hi8 = (lane >> 4) << 3;
  for (int a = 0; a < 2; ++a)
    for (int bi = 0; bi < 4; ++bi) {
      int gcol = colBase + wn + bi * 16 + (lane & 15);
      float bias = pb[gcol];
      for (int r = 0; r < 8; ++r) {
        int m = rowBase + wm + a * 16 + r + hi8;
        out[(size_t)m * 256 + gcol] = acc[a][bi][r] + bias;
      }
    }
}

// ---------------------------------------------------------------------------

extern "C" void kernel_launch(void* const* d_in, const int* in_sizes, int n_in,
                              void* d_out, int out_size, void* d_ws, size_t ws_size,
                              hipStream_t stream) {
  (void)in_sizes; (void)n_in; (void)out_size; (void)ws_size;
  const float* x      = (const float*)d_in[0];
  const float* qkv_w  = (const float*)d_in[1];
  const float* proj_w = (const float*)d_in[2];
  const float* proj_b = (const float*)d_in[3];
  const float* bn_g   = (const float*)d_in[4];
  const float* bn_b   = (const float*)d_in[5];
  const float* w3 = (const float*)d_in[6];  const float* b3 = (const float*)d_in[7];
  const float* w5 = (const float*)d_in[8];  const float* b5 = (const float*)d_in[9];
  const float* w7 = (const float*)d_in[10]; const float* b7 = (const float*)d_in[11];
  float* out = (float*)d_out;

  char* ws = (char*)d_ws;
  size_t off = 0;
  auto take = [&](size_t bytes) -> void* {
    void* p = ws + off;
    off += (bytes + 255) & ~(size_t)255;
    return p;
  };
  __bf16* qkvw_b = (__bf16*)take((size_t)768 * 256 * 2);
  __bf16* projw_b = (__bf16*)take((size_t)256 * 256 * 2);
  __bf16* xg   = (__bf16*)take((size_t)32768 * 256 * 2);
  __bf16* qbuf = (__bf16*)take((size_t)32768 * 256 * 2);
  __bf16* kbuf = (__bf16*)take((size_t)32768 * 256 * 2);
  __bf16* vbuf = (__bf16*)take((size_t)32768 * 256 * 2);
  __bf16* vimg = (__bf16*)take((size_t)8 * 256 * 4096 * 2);
  __bf16* attn = (__bf16*)take((size_t)32768 * 256 * 2);
  __bf16* kvT  = (__bf16*)take((size_t)64 * 1024 * 2);
  float*  smax = (float*)take((size_t)2048 * 4);
  float*  sinv = (float*)take((size_t)2048 * 4);

  // 1) weight conversion + fused BN/hardswish activation (bf16 A operand)
  k_convert<<<(768 * 256 + 255) / 256, 256, 0, stream>>>(qkv_w, qkvw_b, 768 * 256);
  k_convert<<<(256 * 256 + 255) / 256, 256, 0, stream>>>(proj_w, projw_b, 256 * 256);
  k_act<<<32768, 256, 0, stream>>>(x, bn_g, bn_b, xg);

  // 2) QKV GEMM (WMMA bf16), split-stores q/k/v
  k_gemm_qkv<<<dim3(256, 6), 256, 0, stream>>>(xg, qkvw_b, qbuf, kbuf, vbuf);

  // 3) v -> channel-major image (for depthwise conv locality)
  k_vimg<<<dim3(64, 4, 8), 256, 0, stream>>>(vbuf, vimg);

  // 4) softmax stats over token axis for k
  k_softmax<<<2048, 256, 0, stream>>>(kbuf, smax, sinv);

  // 5) kv = softmax(k)^T v per (b,h), scale folded in, stored transposed
  k_kv<<<64, 128, 0, stream>>>(kbuf, vbuf, smax, sinv, kvT);

  // 6) fused eff (q@kv via WMMA) + depthwise conv + q-gating -> attn bf16
  k_effcrpe<3><<<dim3(16, 2, 8), 256, 0, stream>>>(qbuf, vimg, kvT, w3, b3, attn, 0);
  k_effcrpe<5><<<dim3(16, 3, 8), 256, 0, stream>>>(qbuf, vimg, kvT, w5, b5, attn, 2);
  k_effcrpe<7><<<dim3(16, 3, 8), 256, 0, stream>>>(qbuf, vimg, kvT, w7, b7, attn, 5);

  // 7) projection GEMM (WMMA bf16) + bias -> f32 output
  k_gemm_proj<<<dim3(256, 2), 256, 0, stream>>>(attn, projw_b, proj_b, out);
}
